// SPELL_25125558682278
// MI455X (gfx1250) — matrix operand order repaired and checked by reading, not compiled, not run on hardware
//
#include <hip/hip_runtime.h>
#include <math.h>

// ---------------------------------------------------------------------------
// GNN pipeline for MI455X (gfx1250, wave32).  All dense GEMMs run on
// v_wmma_f32_16x16x32_f16 (f32 accumulate).  Weights are pre-converted once
// per launch to transposed f16 (contiguous-K) so B fragments are two b128
// loads; activations feeding GEMMs are kept in f16 so A fragments are b128
// loads with no inner-loop converts.  The EdgeConv gather is fused into the
// A-fragment loader with the concat-half chosen at compile time (branch-free).
// Segment-max / segment-sum use global atomics.
// ---------------------------------------------------------------------------

typedef __attribute__((ext_vector_type(16))) _Float16 v16h;
typedef __attribute__((ext_vector_type(8)))  float    v8f;

#define NEGF (-1.0e30f)

// ---------------- WMMA fragment loaders (ISA 7.12.2 layouts) ---------------
// 16-bit A matrix 16x32 (MxK): lane L holds row M = L&15; half = (L>>4)&1.
// Element e: vgpr v=e>>1, k = (v<4 ? 2v : 16+2(v-4)) + (e&1) + 8*half.
// => within a lane, elements 0..7 are k = base+0..7, elements 8..15 are
// k = base+16..23 : two contiguous 8-element runs (16B each -> b128 loads).
__device__ inline v16h frag_a(const _Float16* __restrict__ A, int lda,
                              int row, int k0, int half) {
  const _Float16* r = A + (size_t)row * lda + k0 + 8 * half;
  v16h a;
#pragma unroll
  for (int i = 0; i < 8; ++i) a[i] = r[i];
#pragma unroll
  for (int i = 0; i < 8; ++i) a[8 + i] = r[16 + i];
  return a;
}
__device__ inline v16h frag_a(const float* __restrict__ A, int lda,
                              int row, int k0, int half) {
  const float* r = A + (size_t)row * lda + k0 + 8 * half;
  v16h a;
#pragma unroll
  for (int i = 0; i < 8; ++i) a[i] = (_Float16)r[i];
#pragma unroll
  for (int i = 0; i < 8; ++i) a[8 + i] = (_Float16)r[16 + i];
  return a;
}

// B fragment from transposed f16 weights WT[n][k] (ld = K): lane L holds
// column n = L&15; needs k = k0+8*half+0..7 and k0+16+8*half+0..7 (two 16B runs).
__device__ inline v16h frag_b_t(const _Float16* __restrict__ BT, int K,
                                int k0, int n, int half) {
  const _Float16* c = BT + (size_t)n * K + k0 + 8 * half;
  v16h b;
#pragma unroll
  for (int e = 0; e < 16; ++e) b[e] = c[e < 8 ? e : e + 8];
  return b;
}

__device__ inline v8f wmma_f16(v16h a, v16h b, v8f c) {
  return __builtin_amdgcn_wmma_f32_16x16x32_f16(false, a, false, b,
                                                (short)0, c, false, false);
}

__device__ inline bool maskok(int a, int mode) {
  // mode 0: attr<=0 (forward), 1: attr>=0 (backward), 2: all
  return (mode == 2) || (mode == 0 ? (a <= 0) : (a >= 0));
}

__device__ inline void atomicMaxFloat(float* addr, float val) {
  if (val >= 0.0f) atomicMax((int*)addr, __float_as_int(val));
  else             atomicMin((unsigned int*)addr, (unsigned int)__float_as_int(val));
}

// ---------------- generic WMMA GEMM:  C = A@B (+bias)  or  C += ... --------
// One wave per block; blockIdx.x -> 16-row strip, blockIdx.y -> group of NT
// 16-column tiles; A fragment reused across the NT tiles (back-to-back WMMA).
template <int NT, bool ACC, typename TA>
__global__ void k_gemm(const TA* __restrict__ A, const _Float16* __restrict__ BT,
                       const float* __restrict__ bias, float* __restrict__ C,
                       int M, int K, int Nc) {
  const int lane = threadIdx.x;
  const int half = (lane >> 4) & 1;
  const int nlo  = lane & 15;
  const int m0   = blockIdx.x * 16;
  const int n0   = blockIdx.y * (16 * NT);
  const int row_a = min(m0 + nlo, M - 1);

  v8f acc[NT];
#pragma unroll
  for (int t = 0; t < NT; ++t) acc[t] = (v8f){0.f,0.f,0.f,0.f,0.f,0.f,0.f,0.f};

  for (int k0 = 0; k0 < K; k0 += 32) {
    v16h a = frag_a(A, K, row_a, k0, half);
#pragma unroll
    for (int t = 0; t < NT; ++t) {
      v16h b = frag_b_t(BT, K, k0, n0 + 16 * t + nlo, half);
      acc[t] = wmma_f16(a, b, acc[t]);
    }
  }
#pragma unroll
  for (int t = 0; t < NT; ++t) {
    int n = n0 + 16 * t + nlo;
    float bv = bias ? bias[n] : 0.f;
#pragma unroll
    for (int j = 0; j < 8; ++j) {
      int row = m0 + j + 8 * half;      // C/D layout: M = j + 8*half
      if (row < M) {
        size_t idx = (size_t)row * Nc + n;
        float v = acc[t][j] + bv;
        if (ACC) C[idx] += v; else C[idx] = v;
      }
    }
  }
}

// ---------------- EdgeConv layer 1: fused gather + GEMM + ReLU -------------
// A row e is [ h[dst] (k=0..63) , h[src]-h[dst] (k=64..127) ].  The two
// 64-wide halves are handled by separate branch-free loaders so every
// fragment is two contiguous 8xf16 runs (vectorizable) per source row.
__device__ inline v16h frag_a_ec_lo(const _Float16* __restrict__ hd,
                                    int k0, int half) {      // k0 in {0,32}
  const _Float16* r = hd + k0 + 8 * half;
  v16h a;
#pragma unroll
  for (int i = 0; i < 8; ++i) a[i] = r[i];
#pragma unroll
  for (int i = 0; i < 8; ++i) a[8 + i] = r[16 + i];
  return a;
}
__device__ inline v16h frag_a_ec_hi(const _Float16* __restrict__ hs,
                                    const _Float16* __restrict__ hd,
                                    int k0, int half) {      // k0 in {0,32}
  const _Float16* rs = hs + k0 + 8 * half;
  const _Float16* rd = hd + k0 + 8 * half;
  v16h a;
#pragma unroll
  for (int i = 0; i < 8; ++i) a[i] = (_Float16)(rs[i] - rd[i]);
#pragma unroll
  for (int i = 0; i < 8; ++i) a[8 + i] = (_Float16)(rs[16 + i] - rd[16 + i]);
  return a;
}

__global__ void k_ec_gemm1(const _Float16* __restrict__ h, const int* __restrict__ ei,
                           const _Float16* __restrict__ W1T, const float* __restrict__ b1,
                           _Float16* __restrict__ m1, int E) {
  const int lane = threadIdx.x;
  const int half = (lane >> 4) & 1;
  const int nlo  = lane & 15;
  const int e0   = blockIdx.x * 16;
  const int me   = min(e0 + nlo, E - 1);
  const _Float16* hs = h + (size_t)ei[me] * 64;
  const _Float16* hd = h + (size_t)ei[E + me] * 64;

  v8f acc[4];
#pragma unroll
  for (int t = 0; t < 4; ++t) acc[t] = (v8f){0.f,0.f,0.f,0.f,0.f,0.f,0.f,0.f};

#pragma unroll
  for (int kk = 0; kk < 4; ++kk) {           // K = 128; halves chosen statically
    v16h a = (kk < 2) ? frag_a_ec_lo(hd, kk * 32, half)
                      : frag_a_ec_hi(hs, hd, (kk - 2) * 32, half);
#pragma unroll
    for (int t = 0; t < 4; ++t) {
      v16h b = frag_b_t(W1T, 128, kk * 32, 16 * t + nlo, half);
      acc[t] = wmma_f16(a, b, acc[t]);
    }
  }
#pragma unroll
  for (int t = 0; t < 4; ++t) {
    int n = 16 * t + nlo;
    float bv = b1[n];
#pragma unroll
    for (int j = 0; j < 8; ++j) {
      int row = e0 + j + 8 * half;
      if (row < E) m1[(size_t)row * 64 + n] = (_Float16)fmaxf(acc[t][j] + bv, 0.f);
    }
  }
}

// ---------------- EdgeConv layer 2: GEMM + masked atomic-max scatter -------
__global__ void k_ec_gemm2(const _Float16* __restrict__ m1, const _Float16* __restrict__ W2T,
                           const float* __restrict__ b2, const int* __restrict__ ei,
                           const int* __restrict__ attr, int mode,
                           float* __restrict__ ec, int E) {
  const int lane = threadIdx.x;
  const int half = (lane >> 4) & 1;
  const int nlo  = lane & 15;
  const int e0   = blockIdx.x * 16;
  const int row_a = min(e0 + nlo, E - 1);

  v8f acc[4];
#pragma unroll
  for (int t = 0; t < 4; ++t) acc[t] = (v8f){0.f,0.f,0.f,0.f,0.f,0.f,0.f,0.f};

#pragma unroll
  for (int kk = 0; kk < 2; ++kk) {           // K = 64
    v16h a = frag_a(m1, 64, row_a, kk * 32, half);
#pragma unroll
    for (int t = 0; t < 4; ++t) {
      v16h b = frag_b_t(W2T, 64, kk * 32, 16 * t + nlo, half);
      acc[t] = wmma_f16(a, b, acc[t]);
    }
  }
#pragma unroll
  for (int t = 0; t < 4; ++t) {
    int n = 16 * t + nlo;
    float bv = b2[n];
#pragma unroll
    for (int j = 0; j < 8; ++j) {
      int e = e0 + j + 8 * half;
      if (e < E && maskok(attr[e], mode)) {
        int d = ei[E + e];
        atomicMaxFloat(&ec[(size_t)d * 64 + n], acc[t][j] + bv);
      }
    }
  }
}

// ---------------- elementwise / reduction helpers --------------------------
__global__ void k_fill(float* p, float v, long long n) {
  long long i = (long long)blockIdx.x * blockDim.x + threadIdx.x;
  if (i < n) p[i] = v;
}
__global__ void k_maxfix(float* p, long long n) {
  long long i = (long long)blockIdx.x * blockDim.x + threadIdx.x;
  if (i < n) p[i] = (p[i] > -5.0e29f) ? p[i] : 0.f;
}
// transpose + f16 convert weights: WT[n*K+k] = W[k*Nc+n]
__global__ void k_wT(const float* __restrict__ W, _Float16* __restrict__ WT,
                     int K, int Nc) {
  int idx = blockIdx.x * blockDim.x + threadIdx.x;
  if (idx >= K * Nc) return;
  int k = idx / Nc, n = idx % Nc;
  WT[(size_t)n * K + k] = (_Float16)W[idx];
}
// column sums + sumsq (population stats) via atomics into stats[0:C],[C:2C]
__global__ void k_colstats(const float* __restrict__ y, int n_rows, int C,
                           float* __restrict__ stats) {
  int tid = threadIdx.x;
  int G   = blockDim.x / C;           // 256/C groups (C = 64 or 256)
  int col = tid % C;
  int g   = tid / C;
  int r0  = blockIdx.x * 128 + g;
  int r1  = min(n_rows, blockIdx.x * 128 + 128);
  float s = 0.f, ss = 0.f;
  for (int r = r0; r < r1; r += G) {
    float v = y[(size_t)r * C + col];
    s += v; ss += v * v;
  }
  atomicAdd(&stats[col], s);
  atomicAdd(&stats[C + col], ss);
}
// BN + ReLU in place; optionally also emit an f16 copy for downstream WMMA
__global__ void k_bn_relu(float* __restrict__ y, const float* __restrict__ stats,
                          const float* __restrict__ g, const float* __restrict__ b,
                          int n_rows, int C, _Float16* __restrict__ y16) {
  long long idx = (long long)blockIdx.x * blockDim.x + threadIdx.x;
  long long total = (long long)n_rows * C;
  if (idx >= total) return;
  int col = (int)(idx % C);
  float mu  = stats[col] / (float)n_rows;
  float var = stats[C + col] / (float)n_rows - mu * mu;
  float v = (y[idx] - mu) * rsqrtf(var + 1e-5f) * g[col] + b[col];
  v = fmaxf(v, 0.f);
  y[idx] = v;
  if (y16) y16[idx] = (_Float16)v;
}

// ---------------- GATv2 edge kernels ---------------------------------------
__global__ void k_gat_score(const float* __restrict__ xl, const float* __restrict__ xr,
                            const float* __restrict__ att, const int* __restrict__ ei,
                            const int* __restrict__ attr, int mode,
                            float* __restrict__ esc, float* __restrict__ emax, int E) {
  long long idx = (long long)blockIdx.x * blockDim.x + threadIdx.x;
  if (idx >= (long long)E * 4) return;
  int e = (int)(idx >> 2), hh = (int)(idx & 3);
  int s = ei[e], d = ei[E + e];
  const float* pl = xl + (size_t)s * 256 + hh * 64;
  const float* pr = xr + (size_t)d * 256 + hh * 64;
  const float* pa = att + hh * 64;
  float acc = 0.f;
#pragma unroll 4
  for (int c = 0; c < 64; ++c) {
    float v = pl[c] + pr[c];
    v = (v > 0.f) ? v : 0.2f * v;       // leaky_relu(0.2)
    acc += v * pa[c];
  }
  esc[idx] = acc;
  if (maskok(attr[e], mode)) atomicMaxFloat(&emax[(size_t)d * 4 + hh], acc);
}
__global__ void k_gat_den(const float* __restrict__ esc, const float* __restrict__ emax,
                          const int* __restrict__ ei, const int* __restrict__ attr,
                          int mode, float* __restrict__ den, int E) {
  long long idx = (long long)blockIdx.x * blockDim.x + threadIdx.x;
  if (idx >= (long long)E * 4) return;
  int e = (int)(idx >> 2), hh = (int)(idx & 3);
  if (!maskok(attr[e], mode)) return;
  int d = ei[E + e];
  atomicAdd(&den[(size_t)d * 4 + hh], __expf(esc[idx] - emax[(size_t)d * 4 + hh]));
}
__global__ void k_gat_wsum(const float* __restrict__ esc, const float* __restrict__ emax,
                           const float* __restrict__ xl, const int* __restrict__ ei,
                           const int* __restrict__ attr, int mode,
                           float* __restrict__ gout, int E) {
  long long idx = (long long)blockIdx.x * blockDim.x + threadIdx.x;
  if (idx >= (long long)E * 256) return;
  int e  = (int)(idx >> 8);
  int ch = (int)(idx & 255);
  int hh = ch >> 6;
  if (!maskok(attr[e], mode)) return;
  int s = ei[e], d = ei[E + e];
  float w = __expf(esc[(size_t)e * 4 + hh] - emax[(size_t)d * 4 + hh]);
  atomicAdd(&gout[(size_t)d * 256 + ch], w * xl[(size_t)s * 256 + ch]);
}
__global__ void k_gat_fin(float* __restrict__ gout, const float* __restrict__ den,
                          const float* __restrict__ gbias, int n_rows) {
  long long idx = (long long)blockIdx.x * blockDim.x + threadIdx.x;
  if (idx >= (long long)n_rows * 256) return;
  int ch = (int)(idx & 255);
  int hh = ch >> 6;
  int row = (int)(idx >> 8);
  gout[idx] = gout[idx] / fmaxf(den[(size_t)row * 4 + hh], 1e-16f) + gbias[ch];
}

// ---------------- SAGE aggregation -----------------------------------------
__global__ void k_sage_cnt(const int* __restrict__ ei, const int* __restrict__ attr,
                           int mode, float* __restrict__ cnt, int E) {
  int e = blockIdx.x * blockDim.x + threadIdx.x;
  if (e >= E || !maskok(attr[e], mode)) return;
  atomicAdd(&cnt[ei[E + e]], 1.f);
}
__global__ void k_sage_scatter(const float* __restrict__ z2, const int* __restrict__ ei,
                               const int* __restrict__ attr, int mode,
                               float* __restrict__ s, int E) {
  long long idx = (long long)blockIdx.x * blockDim.x + threadIdx.x;
  if (idx >= (long long)E * 256) return;
  int e  = (int)(idx >> 8);
  int ch = (int)(idx & 255);
  if (!maskok(attr[e], mode)) return;
  atomicAdd(&s[(size_t)ei[E + e] * 256 + ch], z2[(size_t)ei[e] * 256 + ch]);
}
__global__ void k_sage_mean(const float* __restrict__ s, const float* __restrict__ cnt,
                            _Float16* __restrict__ mean16, int n_rows) {
  long long idx = (long long)blockIdx.x * blockDim.x + threadIdx.x;
  if (idx >= (long long)n_rows * 256) return;
  mean16[idx] = (_Float16)(s[idx] / fmaxf(cnt[idx >> 8], 1.f));
}

// ---------------------------------------------------------------------------
static inline unsigned blk1d(long long n) { return (unsigned)((n + 255) / 256); }

extern "C" void kernel_launch(void* const* d_in, const int* in_sizes, int n_in,
                              void* d_out, int out_size, void* d_ws, size_t ws_size,
                              hipStream_t stream) {
  const int D_IN = 1024, C1 = 64, HC = 256, FINAL = 32;
  const int Nn = in_sizes[0] / D_IN;      // 50000
  const int E  = in_sizes[2];             // 300000

  const float* x    = (const float*)d_in[0];
  const int*   ei   = (const int*)d_in[1];     // [2,E] row-major
  const int*   attr = (const int*)d_in[2];
  const float* W011 = (const float*)d_in[3];
  const float* b011 = (const float*)d_in[4];
  const float* bn01g = (const float*)d_in[5];
  const float* bn01b = (const float*)d_in[6];
  const float* ecW1[3], *ecb1[3], *ecW2[3], *ecb2[3], *ecg[3], *ecb[3];
  for (int i = 0; i < 3; ++i) {
    int base = 7 + 6 * i;
    ecW1[i] = (const float*)d_in[base + 0];
    ecb1[i] = (const float*)d_in[base + 1];
    ecW2[i] = (const float*)d_in[base + 2];
    ecb2[i] = (const float*)d_in[base + 3];
    ecg[i]  = (const float*)d_in[base + 4];
    ecb[i]  = (const float*)d_in[base + 5];
  }
  const float* gWl   = (const float*)d_in[25];
  const float* gbl   = (const float*)d_in[26];
  const float* gWr   = (const float*)d_in[27];
  const float* gbr   = (const float*)d_in[28];
  const float* gatt  = (const float*)d_in[29];
  const float* gbias = (const float*)d_in[30];
  const float* bn21g = (const float*)d_in[31];
  const float* bn21b = (const float*)d_in[32];
  const float* sWl[3], *sbl[3], *sWr[3];
  for (int i = 0; i < 3; ++i) {
    int base = 33 + 3 * i;
    sWl[i] = (const float*)d_in[base + 0];
    sbl[i] = (const float*)d_in[base + 1];
    sWr[i] = (const float*)d_in[base + 2];
  }
  float* out = (float*)d_out;

  // ---- workspace layout, ~257 MB with phase-disjoint region reuse ---------
  char* base = (char*)d_ws;
  size_t o = 0;
  auto alloc = [&](size_t bytes) -> char* {
    char* p = base + o; o = (o + bytes + 255) & ~(size_t)255; return p;
  };
  float*     H     = (float*)alloc((size_t)Nn * C1 * 4);   // y0 / h (BN in place)
  _Float16*  Hh    = (_Float16*)alloc((size_t)Nn * C1 * 2);
  char*      M1r   = alloc((size_t)E * C1 * 2);            // 38.4 MB region
  _Float16*  M1    = (_Float16*)M1r;                       // EdgeConv hidden (f16)
  _Float16*  MEANh = (_Float16*)M1r;                       // later: SAGE mean (25.6 MB)
  float*     ESC   = (float*)(M1r + (size_t)Nn * HC * 2);  // later: edge scores (4.8 MB)
  float*     EMAX  = ESC + (size_t)E * 4;
  float*     DEN   = EMAX + (size_t)Nn * 4;
  float*     EC    = (float*)alloc((size_t)Nn * C1 * 4);   // EdgeConv out / z
  _Float16*  Zh    = (_Float16*)alloc((size_t)Nn * C1 * 2);
  float*     XL    = (float*)alloc((size_t)Nn * HC * 4);   // GAT left; later SAGE sums
  float*     S     = XL;
  float*     XR    = (float*)alloc((size_t)Nn * HC * 4);
  float*     GOUT  = (float*)alloc((size_t)Nn * HC * 4);   // GAT out / z2
  _Float16*  Z2h   = (_Float16*)alloc((size_t)Nn * HC * 2);
  float*     CNT   = (float*)alloc((size_t)Nn * 4);
  float*     ST    = (float*)alloc(2048);
  // transposed f16 weights
  _Float16* W011T = (_Float16*)alloc((size_t)D_IN * C1 * 2);
  _Float16 *ecW1T[3], *ecW2T[3], *sWlT[3], *sWrT[3];
  for (int i = 0; i < 3; ++i) {
    ecW1T[i] = (_Float16*)alloc((size_t)2 * C1 * C1 * 2);
    ecW2T[i] = (_Float16*)alloc((size_t)C1 * C1 * 2);
    sWlT[i]  = (_Float16*)alloc((size_t)HC * FINAL * 2);
    sWrT[i]  = (_Float16*)alloc((size_t)HC * FINAL * 2);
  }
  _Float16* gWlT = (_Float16*)alloc((size_t)C1 * HC * 2);
  _Float16* gWrT = (_Float16*)alloc((size_t)C1 * HC * 2);
  (void)ws_size; (void)n_in; (void)out_size;

  // ---- weight conversion (transpose + f16), once per launch ---------------
  k_wT<<<blk1d((long long)D_IN * C1), 256, 0, stream>>>(W011, W011T, D_IN, C1);
  for (int i = 0; i < 3; ++i) {
    k_wT<<<blk1d((long long)2 * C1 * C1), 256, 0, stream>>>(ecW1[i], ecW1T[i], 2 * C1, C1);
    k_wT<<<blk1d((long long)C1 * C1), 256, 0, stream>>>(ecW2[i], ecW2T[i], C1, C1);
    k_wT<<<blk1d((long long)HC * FINAL), 256, 0, stream>>>(sWl[i], sWlT[i], HC, FINAL);
    k_wT<<<blk1d((long long)HC * FINAL), 256, 0, stream>>>(sWr[i], sWrT[i], HC, FINAL);
  }
  k_wT<<<blk1d((long long)C1 * HC), 256, 0, stream>>>(gWl, gWlT, C1, HC);
  k_wT<<<blk1d((long long)C1 * HC), 256, 0, stream>>>(gWr, gWrT, C1, HC);

  // ---- output accumulator must start at zero every call -------------------
  k_fill<<<blk1d((long long)Nn * FINAL), 256, 0, stream>>>(out, 0.f, (long long)Nn * FINAL);

  // ---- stage 0: h = relu(BN(x @ W011 + b011)) -----------------------------
  k_gemm<4, false><<<dim3(Nn / 16, 1), 32, 0, stream>>>(x, W011T, b011, H, Nn, D_IN, C1);
  k_fill<<<1, 256, 0, stream>>>(ST, 0.f, 128);
  k_colstats<<<(Nn + 127) / 128, 256, 0, stream>>>(H, Nn, C1, ST);
  k_bn_relu<<<blk1d((long long)Nn * C1), 256, 0, stream>>>(H, ST, bn01g, bn01b, Nn, C1, Hh);

  for (int br = 0; br < 3; ++br) {
    const int mode = br;  // 0: attr<=0, 1: attr>=0, 2: all

    // ---- EdgeConv: m1 = relu([xi, xj-xi]@W1+b1); max-scatter of m1@W2+b2 --
    k_ec_gemm1<<<E / 16, 32, 0, stream>>>(Hh, ei, ecW1T[br], ecb1[br], M1, E);
    k_fill<<<blk1d((long long)Nn * C1), 256, 0, stream>>>(EC, NEGF, (long long)Nn * C1);
    k_ec_gemm2<<<E / 16, 32, 0, stream>>>(M1, ecW2T[br], ecb2[br], ei, attr, mode, EC, E);
    k_maxfix<<<blk1d((long long)Nn * C1), 256, 0, stream>>>(EC, (long long)Nn * C1);
    k_fill<<<1, 256, 0, stream>>>(ST, 0.f, 128);
    k_colstats<<<(Nn + 127) / 128, 256, 0, stream>>>(EC, Nn, C1, ST);
    k_bn_relu<<<blk1d((long long)Nn * C1), 256, 0, stream>>>(EC, ST, ecg[br], ecb[br], Nn, C1, Zh);
    // Zh now holds z [N,64] (f16)

    // ---- GATv2 ------------------------------------------------------------
    k_gemm<4, false><<<dim3(Nn / 16, 4), 32, 0, stream>>>(Zh, gWlT, gbl, XL, Nn, C1, HC);
    k_gemm<4, false><<<dim3(Nn / 16, 4), 32, 0, stream>>>(Zh, gWrT, gbr, XR, Nn, C1, HC);
    k_fill<<<blk1d((long long)Nn * 4), 256, 0, stream>>>(EMAX, NEGF, (long long)Nn * 4);
    k_gat_score<<<blk1d((long long)E * 4), 256, 0, stream>>>(XL, XR, gatt, ei, attr, mode, ESC, EMAX, E);
    k_maxfix<<<blk1d((long long)Nn * 4), 256, 0, stream>>>(EMAX, (long long)Nn * 4);
    k_fill<<<blk1d((long long)Nn * 4), 256, 0, stream>>>(DEN, 0.f, (long long)Nn * 4);
    k_gat_den<<<blk1d((long long)E * 4), 256, 0, stream>>>(ESC, EMAX, ei, attr, mode, DEN, E);
    k_fill<<<blk1d((long long)Nn * HC), 256, 0, stream>>>(GOUT, 0.f, (long long)Nn * HC);
    k_gat_wsum<<<blk1d((long long)E * HC), 256, 0, stream>>>(ESC, EMAX, XL, ei, attr, mode, GOUT, E);
    k_gat_fin<<<blk1d((long long)Nn * HC), 256, 0, stream>>>(GOUT, DEN, gbias, Nn);
    k_fill<<<1, 256, 0, stream>>>(ST, 0.f, 512);
    k_colstats<<<(Nn + 127) / 128, 256, 0, stream>>>(GOUT, Nn, HC, ST);
    k_bn_relu<<<blk1d((long long)Nn * HC), 256, 0, stream>>>(GOUT, ST, bn21g, bn21b, Nn, HC, Z2h);
    // GOUT = z2 (f32), Z2h = z2 (f16)

    // ---- SAGE: out += mean(z2[src]) @ Wl + bl + z2 @ Wr --------------------
    k_fill<<<blk1d((long long)Nn * HC), 256, 0, stream>>>(S, 0.f, (long long)Nn * HC);
    k_fill<<<blk1d((long long)Nn), 256, 0, stream>>>(CNT, 0.f, (long long)Nn);
    k_sage_cnt<<<blk1d((long long)E), 256, 0, stream>>>(ei, attr, mode, CNT, E);
    k_sage_scatter<<<blk1d((long long)E * HC), 256, 0, stream>>>(GOUT, ei, attr, mode, S, E);
    k_sage_mean<<<blk1d((long long)Nn * HC), 256, 0, stream>>>(S, CNT, MEANh, Nn);
    k_gemm<2, true><<<dim3(Nn / 16, 1), 32, 0, stream>>>(MEANh, sWlT[br], sbl[br], out, Nn, HC, FINAL);
    k_gemm<2, true><<<dim3(Nn / 16, 1), 32, 0, stream>>>(Z2h,   sWrT[br], nullptr,  out, Nn, HC, FINAL);
  }
}
